// LanguageModel_22282290331866
// MI455X (gfx1250) — compile-verified, matrix operand order checked by reading, not run on hardware
//
#include <hip/hip_runtime.h>
#include <hip/hip_bf16.h>
#include <math.h>

typedef _Float16 half_t;
typedef __attribute__((ext_vector_type(16))) _Float16 v16h;
typedef __attribute__((ext_vector_type(8)))  _Float16 v8h;
typedef __attribute__((ext_vector_type(8)))  float    v8f;

#define LL 6
#define HH 6
#define DD 64
#define EE 384
#define FF 1536
#define TT 256
#define BBATCH 256
#define VV 65
#define BT (BBATCH * TT)

// ---------------------------------------------------------------------------
// WMMA helpers (CDNA5 16x16x32 f16 -> f32)
// ---------------------------------------------------------------------------
__device__ inline v8f wmma32(v16h a, v16h b, v8f c) {
  return __builtin_amdgcn_wmma_f32_16x16x32_f16(
      /*neg_a=*/false, a, /*neg_b=*/false, b,
      /*c_mod=*/(short)0, c, /*reuse_a=*/false, /*reuse_b=*/false);
}

// Two 16B loads -> one fragment register set: p[0..7] -> [0..7], p[16..23] -> [8..15]
__device__ inline v16h frag_rows16(const half_t* p) {
  v8h lo = *reinterpret_cast<const v8h*>(p);
  v8h hi = *reinterpret_cast<const v8h*>(p + 16);
  v16h a;
#pragma unroll
  for (int i = 0; i < 8; ++i) { a[i] = lo[i]; a[i + 8] = hi[i]; }
  return a;
}

// A fragment: 16x32 (MxK) from row-major source, leading dim lda (halves).
// ISA layout: lanes 0-15 row M=lane, K=0..7 & 16..23; lanes 16-31 K=8..15 & 24..31.
// Also used for B = X^T fragments when X is row-major [N,K] (e.g. K-matrix, V^T).
__device__ inline v16h load_a(const half_t* A, int lda) {
  int lane = threadIdx.x & 31;
  int row  = lane & 15;
  int kb   = (lane & 16) ? 8 : 0;
  return frag_rows16(A + (size_t)row * lda + kb);
}

// B fragment from pre-packed tile: 32 lanes x 16 contiguous halves.
__device__ inline v16h load_packed(const half_t* tile) {
  const half_t* q = tile + (threadIdx.x & 31) * 16;
  v8h lo = *reinterpret_cast<const v8h*>(q);
  v8h hi = *reinterpret_cast<const v8h*>(q + 8);
  v16h b;
#pragma unroll
  for (int i = 0; i < 8; ++i) { b[i] = lo[i]; b[i + 8] = hi[i]; }
  return b;
}

// Store one 16x16 f32 accumulator fragment with fused bias/residual/relu.
__device__ inline void store_acc(v8f c, float* Cf, half_t* Ch, const float* bias,
                                 const float* resid, int relu,
                                 int row0, int col0, int N) {
  int lane = threadIdx.x & 31;
  int col  = col0 + (lane & 15);
  if (col >= N) return;
  int r0 = row0 + ((lane & 16) ? 8 : 0);
  float bv = bias ? bias[col] : 0.0f;
#pragma unroll
  for (int r = 0; r < 8; ++r) {
    size_t idx = (size_t)(r0 + r) * N + col;
    float v = c[r] + bv;
    if (resid) v += resid[idx];
    if (relu)  v = fmaxf(v, 0.0f);
    if (Cf) Cf[idx] = v;
    if (Ch) Ch[idx] = (half_t)v;
  }
}

// ---------------------------------------------------------------------------
// Generic GEMM: C[M,N] = A[M,K] @ B[K,N] (+bias +residual, opt relu)
// A f16 row-major (lda=K); B pre-packed in WMMA fragment tiles (see pack_b).
// Block: 256 thr = 8 waves; block tile 128(M) x 128(N); wave tile 32x64;
// K step 32. M % 128 == 0 (M = 65536 everywhere). Packed B covers
// Npad = gridDim.x*128 columns (zero-padded past N).
// ---------------------------------------------------------------------------
__global__ __launch_bounds__(256) void gemm_f16(
    const half_t* __restrict__ A, const half_t* __restrict__ Bp,
    const float* __restrict__ bias, const float* resid,
    float* Cf, half_t* Ch, int M, int N, int K, int relu) {
  const int lane  = threadIdx.x & 31;
  const int wave  = threadIdx.x >> 5;
  const int m0    = blockIdx.y * 128 + (wave & 3) * 32;
  const int nbase = blockIdx.x * 128 + (wave >> 2) * 64;
  const int ktiles = K >> 5;
  const size_t fstride = (size_t)ktiles * 512;  // halves per column-frag strip
  v8f acc[2][4] = {};
  const half_t* bp0   = Bp + (size_t)(nbase >> 4) * fstride + lane * 16;
  const half_t* arow0 = A + (size_t)m0 * K;
  const half_t* arow1 = A + (size_t)(m0 + 16) * K;
  for (int kt = 0; kt < ktiles; ++kt) {
    v16h a0 = load_a(arow0 + kt * 32, K);
    v16h a1 = load_a(arow1 + kt * 32, K);
    const half_t* bq = bp0 + (size_t)kt * 512;
#pragma unroll
    for (int nf = 0; nf < 4; ++nf) {
      v8h lo = *reinterpret_cast<const v8h*>(bq + (size_t)nf * fstride);
      v8h hi = *reinterpret_cast<const v8h*>(bq + (size_t)nf * fstride + 8);
      v16h b;
#pragma unroll
      for (int i = 0; i < 8; ++i) { b[i] = lo[i]; b[i + 8] = hi[i]; }
      acc[0][nf] = wmma32(a0, b, acc[0][nf]);
      acc[1][nf] = wmma32(a1, b, acc[1][nf]);
    }
  }
#pragma unroll
  for (int mi = 0; mi < 2; ++mi)
#pragma unroll
    for (int nf = 0; nf < 4; ++nf)
      store_acc(acc[mi][nf], Cf, Ch, bias, resid, relu,
                m0 + mi * 16, nbase + nf * 16, N);
}

// ---------------------------------------------------------------------------
// Fused causal attention for one (batch, head).
// LDS: S scores 256x256 f32 (256KB) + V^T 64x256 f16 (32KB) = 288KB <= 320KB.
// S=QK^T via WMMA, causal softmax in f32 (probs repacked f16 in place),
// O = P@V via WMMA with both operands in LDS.
// Q/K/V/O are [T, D] strided views with leading dim E (= H*D).
// ---------------------------------------------------------------------------
__global__ __launch_bounds__(256) void attn_kernel(
    const half_t* __restrict__ Qg, const half_t* __restrict__ Kg,
    const half_t* __restrict__ Vg, half_t* __restrict__ Og) {
  extern __shared__ float smem[];
  half_t* vT = reinterpret_cast<half_t*>(smem + TT * TT);  // [D][T] halves

  const int bh = blockIdx.x;
  const int b = bh / HH, h = bh % HH;
  const size_t base = (size_t)b * TT * EE + (size_t)h * DD;
  const half_t* Q  = Qg + base;
  const half_t* Km = Kg + base;
  const half_t* V  = Vg + base;
  half_t*       O  = Og + base;

  const int lane = threadIdx.x & 31;
  const int wave = threadIdx.x >> 5;
  const int m0   = wave * 32;            // each wave owns 32 query rows
  const int ccol = lane & 15;
  const int crb  = (lane & 16) ? 8 : 0;
  const float scale = 0.125f;            // D^-0.5, D = 64

  // Phase 0: stage V^T into LDS (coalesced 16B global reads)
  for (int c = threadIdx.x; c < TT * (DD / 8); c += 256) {
    int t  = c >> 3;
    int d0 = (c & 7) * 8;
    v8h v = *reinterpret_cast<const v8h*>(V + (size_t)t * EE + d0);
#pragma unroll
    for (int i = 0; i < 8; ++i) vT[(d0 + i) * TT + t] = v[i];
  }

  // Phase 1: S = Q @ K^T (K^T fragment load == A-fragment pattern on row-major K)
  v16h qa00 = load_a(Q + (size_t)m0 * EE, EE);
  v16h qa01 = load_a(Q + (size_t)m0 * EE + 32, EE);
  v16h qa10 = load_a(Q + (size_t)(m0 + 16) * EE, EE);
  v16h qa11 = load_a(Q + (size_t)(m0 + 16) * EE + 32, EE);
  for (int n0 = 0; n0 < TT; n0 += 16) {
    v16h kb0 = load_a(Km + (size_t)n0 * EE, EE);
    v16h kb1 = load_a(Km + (size_t)n0 * EE + 32, EE);
    v8f c0 = {}, c1 = {};
    c0 = wmma32(qa00, kb0, c0);
    c0 = wmma32(qa01, kb1, c0);
    c1 = wmma32(qa10, kb0, c1);
    c1 = wmma32(qa11, kb1, c1);
    int col = n0 + ccol;
#pragma unroll
    for (int r = 0; r < 8; ++r) {
      int row0 = m0 + crb + r;
      smem[row0 * TT + col] = (col <= row0) ? c0[r] * scale : -__builtin_inff();
      int row1 = row0 + 16;
      smem[row1 * TT + col] = (col <= row1) ? c1[r] * scale : -__builtin_inff();
    }
  }
  __syncthreads();

  // Phase 2: softmax; one thread per row; write f16 probs over the row prefix
  {
    const int row = threadIdx.x;
    float* s = smem + row * TT;
    float mx = -__builtin_inff();
    for (int j = 0; j <= row; ++j) mx = fmaxf(mx, s[j]);
    float sum = 0.0f;
    for (int j = 0; j <= row; ++j) sum += __expf(s[j] - mx);
    float rs = 1.0f / sum;
    half_t* p = reinterpret_cast<half_t*>(s);  // half j clobbers float j/2 (<= j): safe
    for (int j = 0; j < TT; ++j) {
      float e = (j <= row) ? __expf(s[j] - mx) * rs : 0.0f;
      p[j] = (half_t)e;
    }
  }
  __syncthreads();

  // Phase 3: O = P @ V; P rows at half-stride 512 in LDS, V^T fragments from LDS
  const half_t* P = reinterpret_cast<const half_t*>(smem);
  v8f oa[2][4] = {};
  for (int k0 = 0; k0 < TT; k0 += 32) {
    v16h pa0 = load_a(P + (size_t)m0 * 512 + k0, 512);
    v16h pa1 = load_a(P + (size_t)(m0 + 16) * 512 + k0, 512);
#pragma unroll
    for (int nf = 0; nf < 4; ++nf) {
      v16h vb = load_a(vT + (size_t)(nf * 16) * TT + k0, TT);
      oa[0][nf] = wmma32(pa0, vb, oa[0][nf]);
      oa[1][nf] = wmma32(pa1, vb, oa[1][nf]);
    }
  }
#pragma unroll
  for (int mi = 0; mi < 2; ++mi) {
#pragma unroll
    for (int nf = 0; nf < 4; ++nf) {
      int col = nf * 16 + ccol;
#pragma unroll
      for (int r = 0; r < 8; ++r) {
        int row = m0 + mi * 16 + crb + r;
        O[(size_t)row * EE + col] = (half_t)oa[mi][nf][r];
      }
    }
  }
}

// ---------------------------------------------------------------------------
// Weight packing: row-major f32 [K,N] -> f16 WMMA B-fragment tiles.
// Tile (f, kt) covers cols [f*16,f*16+16), K rows [kt*32,kt*32+32).
// Element (tile, s, j): lane s reads halves j=0..15 contiguously in GEMM.
// Layout index: ((f*ktiles + kt)*32 + s)*16 + j. Cols >= N are zero padding.
// ---------------------------------------------------------------------------
__global__ void pack_b_kernel(const float* __restrict__ W, half_t* __restrict__ out,
                              int K, int N, int Npad) {
  size_t i = (size_t)blockIdx.x * 256 + threadIdx.x;
  size_t total = (size_t)Npad * K;
  if (i >= total) return;
  int j = (int)(i & 15);
  int s = (int)((i >> 4) & 31);
  size_t tile = i >> 9;
  int ktiles = K >> 5;
  int kt = (int)(tile % ktiles);
  int f  = (int)(tile / ktiles);
  int col = f * 16 + (s & 15);
  int kb  = (s & 16) ? 8 : 0;
  int k   = kt * 32 + ((j < 8) ? (kb + j) : (8 + kb + j));
  out[i] = (col < N) ? (half_t)W[(size_t)k * N + col] : (half_t)0.0f;
}

// Wq/Wk/Wv: (L,H,E,D) f32 -> per-layer packed B tiles for [E, H*D] (Npad = 384)
__global__ void pack_qkv_kernel(const float* __restrict__ W, half_t* __restrict__ out) {
  size_t i = (size_t)blockIdx.x * 256 + threadIdx.x;
  const size_t per = (size_t)EE * EE;
  if (i >= (size_t)LL * per) return;
  int l = (int)(i / per);
  size_t r = i % per;
  int j = (int)(r & 15);
  int s = (int)((r >> 4) & 31);
  size_t tile = r >> 9;
  const int ktiles = EE >> 5;
  int kt = (int)(tile % ktiles);
  int f  = (int)(tile / ktiles);
  int col = f * 16 + (s & 15);      // h*64 + d
  int kb  = (s & 16) ? 8 : 0;
  int e   = kt * 32 + ((j < 8) ? (kb + j) : (8 + kb + j));
  int h = col / DD, d = col % DD;
  out[i] = (half_t)W[(((size_t)l * HH + h) * EE + e) * DD + d];
}

// ---------------------------------------------------------------------------
// Elementwise / norm / loss kernels
// ---------------------------------------------------------------------------
__global__ void embed_kernel(const int* __restrict__ inp, const float* __restrict__ tok,
                             const float* __restrict__ pos, float* __restrict__ x) {
  size_t i = (size_t)blockIdx.x * 256 + threadIdx.x;
  if (i >= (size_t)BT * EE) return;
  size_t rowt = i / EE;
  int e = (int)(i % EE);
  int t = (int)(rowt % TT);
  x[i] = tok[(size_t)inp[rowt] * EE + e] + pos[(size_t)t * EE + e];
}

__global__ __launch_bounds__(256) void layernorm_kernel(
    const float* __restrict__ x, const float* __restrict__ g,
    const float* __restrict__ bta, half_t* __restrict__ out) {
  int row = blockIdx.x * 8 + (threadIdx.x >> 5);
  int lane = threadIdx.x & 31;
  const float* xr = x + (size_t)row * EE;
  float s = 0.0f, s2 = 0.0f;
  for (int j = lane; j < EE; j += 32) { float v = xr[j]; s += v; s2 += v * v; }
#pragma unroll
  for (int o = 16; o > 0; o >>= 1) { s += __shfl_xor(s, o, 32); s2 += __shfl_xor(s2, o, 32); }
  float mu  = s * (1.0f / EE);
  float var = s2 * (1.0f / EE) - mu * mu;
  float r   = rsqrtf(var + 1e-5f);
  half_t* orow = out + (size_t)row * EE;
  for (int j = lane; j < EE; j += 32)
    orow[j] = (half_t)((xr[j] - mu) * r * g[j] + bta[j]);
}

__global__ __launch_bounds__(256) void loss_rows_kernel(
    const float* __restrict__ logits, const int* __restrict__ targets,
    float* __restrict__ nll) {
  int row = blockIdx.x * 8 + (threadIdx.x >> 5);
  int lane = threadIdx.x & 31;
  const float* lg = logits + (size_t)row * VV;
  float mx = -3.4e38f;
  for (int j = lane; j < VV; j += 32) mx = fmaxf(mx, lg[j]);
#pragma unroll
  for (int o = 16; o > 0; o >>= 1) mx = fmaxf(mx, __shfl_xor(mx, o, 32));
  float s = 0.0f;
  for (int j = lane; j < VV; j += 32) s += __expf(lg[j] - mx);
#pragma unroll
  for (int o = 16; o > 0; o >>= 1) s += __shfl_xor(s, o, 32);
  if (lane == 0) nll[row] = __logf(s) + mx - lg[targets[row]];
}

__global__ __launch_bounds__(256) void loss_reduce_kernel(
    const float* __restrict__ nll, float* __restrict__ out) {
  __shared__ float sh[256];
  float s = 0.0f;
  for (int i = threadIdx.x; i < BT; i += 256) s += nll[i];
  sh[threadIdx.x] = s;
  __syncthreads();
  for (int st = 128; st > 0; st >>= 1) {
    if ((int)threadIdx.x < st) sh[threadIdx.x] += sh[threadIdx.x + st];
    __syncthreads();
  }
  if (threadIdx.x == 0) out[0] = sh[0] * (1.0f / (float)BT);
}

// ---------------------------------------------------------------------------
// Host driver
// ---------------------------------------------------------------------------
extern "C" void kernel_launch(void* const* d_in, const int* in_sizes, int n_in,
                              void* d_out, int out_size, void* d_ws, size_t ws_size,
                              hipStream_t stream) {
  const int*   inputs  = (const int*)  d_in[0];
  const int*   targets = (const int*)  d_in[1];
  const float* tok_emb = (const float*)d_in[2];
  const float* pos_emb = (const float*)d_in[3];
  const float* Wq      = (const float*)d_in[4];
  const float* Wk      = (const float*)d_in[5];
  const float* Wv      = (const float*)d_in[6];
  const float* Wo      = (const float*)d_in[7];
  const float* bo      = (const float*)d_in[8];
  const float* ln1_g   = (const float*)d_in[9];
  const float* ln1_b   = (const float*)d_in[10];
  const float* W1      = (const float*)d_in[11];
  const float* b1      = (const float*)d_in[12];
  const float* W2      = (const float*)d_in[13];
  const float* b2      = (const float*)d_in[14];
  const float* ln2_g   = (const float*)d_in[15];
  const float* ln2_b   = (const float*)d_in[16];
  const float* lnf_g   = (const float*)d_in[17];
  const float* lnf_b   = (const float*)d_in[18];
  const float* Wh      = (const float*)d_in[19];
  const float* bh      = (const float*)d_in[20];
  (void)in_sizes; (void)n_in; (void)out_size; (void)ws_size;

  char* ws = (char*)d_ws;
  size_t off = 0;
  auto take = [&](size_t bytes) -> void* {
    void* p = ws + off;
    off += (bytes + 255) & ~(size_t)255;
    return p;
  };
  float*  x    = (float*) take((size_t)BT * EE * sizeof(float));   // residual stream
  half_t* h16  = (half_t*)take((size_t)BT * EE * 2);               // LN out / attn O (reused)
  half_t* q16  = (half_t*)take((size_t)BT * EE * 2);
  half_t* k16  = (half_t*)take((size_t)BT * EE * 2);
  half_t* vv16 = (half_t*)take((size_t)BT * EE * 2);
  half_t* f116 = (half_t*)take((size_t)BT * FF * 2);               // relu(h@W1+b1)
  half_t* wq16 = (half_t*)take((size_t)LL * EE * EE * 2);          // packed B tiles
  half_t* wk16 = (half_t*)take((size_t)LL * EE * EE * 2);
  half_t* wv16 = (half_t*)take((size_t)LL * EE * EE * 2);
  half_t* wo16 = (half_t*)take((size_t)LL * EE * EE * 2);
  half_t* w116 = (half_t*)take((size_t)LL * EE * FF * 2);
  half_t* w216 = (half_t*)take((size_t)LL * FF * EE * 2);
  half_t* wh16 = (half_t*)take((size_t)EE * 128 * 2);              // Npad = 128
  float*  nll  = (float*) take((size_t)BT * sizeof(float));

  dim3 blk(256);

  // Weight packing (deterministic, every call)
  {
    size_t n = (size_t)LL * EE * EE;
    int g = (int)((n + 255) / 256);
    pack_qkv_kernel<<<g, blk, 0, stream>>>(Wq, wq16);
    pack_qkv_kernel<<<g, blk, 0, stream>>>(Wk, wk16);
    pack_qkv_kernel<<<g, blk, 0, stream>>>(Wv, wv16);
  }
  for (int l = 0; l < LL; ++l) {
    size_t nEE = (size_t)EE * EE, nEF = (size_t)EE * FF;
    pack_b_kernel<<<(int)((nEE + 255) / 256), blk, 0, stream>>>(
        Wo + (size_t)l * EE * EE, wo16 + (size_t)l * nEE, EE, EE, EE);
    pack_b_kernel<<<(int)((nEF + 255) / 256), blk, 0, stream>>>(
        W1 + (size_t)l * EE * FF, w116 + (size_t)l * nEF, EE, FF, FF);
    pack_b_kernel<<<(int)((nEF + 255) / 256), blk, 0, stream>>>(
        W2 + (size_t)l * FF * EE, w216 + (size_t)l * nEF, FF, EE, EE);
  }
  {
    size_t n = (size_t)EE * 128;
    pack_b_kernel<<<(int)((n + 255) / 256), blk, 0, stream>>>(Wh, wh16, EE, VV, 128);
  }

  // Embedding
  {
    size_t n = (size_t)BT * EE;
    embed_kernel<<<(int)((n + 255) / 256), blk, 0, stream>>>(inputs, tok_emb, pos_emb, x);
  }

  const dim3 gE(EE / 128, BT / 128);   // N=384  -> (3, 512)
  const dim3 gF(FF / 128, BT / 128);   // N=1536 -> (12, 512)
  const dim3 gH(1, BT / 128);          // N=65 (Npad=128)
  const size_t attn_lds = (size_t)TT * TT * sizeof(float) + (size_t)DD * TT * 2;

  for (int l = 0; l < LL; ++l) {
    const size_t wEE = (size_t)l * EE * EE;
    layernorm_kernel<<<BT / 8, blk, 0, stream>>>(x, ln1_g + l * EE, ln1_b + l * EE, h16);
    gemm_f16<<<gE, blk, 0, stream>>>(h16, wq16 + wEE, nullptr, nullptr, nullptr, q16,  BT, EE, EE, 0);
    gemm_f16<<<gE, blk, 0, stream>>>(h16, wk16 + wEE, nullptr, nullptr, nullptr, k16,  BT, EE, EE, 0);
    gemm_f16<<<gE, blk, 0, stream>>>(h16, wv16 + wEE, nullptr, nullptr, nullptr, vv16, BT, EE, EE, 0);
    // Attention writes the concatenated head output into h16 (reused as O)
    attn_kernel<<<BBATCH * HH, blk, attn_lds, stream>>>(q16, k16, vv16, h16);
    // x = x + O @ Wo + bo
    gemm_f16<<<gE, blk, 0, stream>>>(h16, wo16 + wEE, bo + l * EE, x, x, nullptr, BT, EE, EE, 0);
    layernorm_kernel<<<BT / 8, blk, 0, stream>>>(x, ln2_g + l * EE, ln2_b + l * EE, h16);
    gemm_f16<<<gF, blk, 0, stream>>>(h16, w116 + (size_t)l * EE * FF, b1 + l * FF,
                                     nullptr, nullptr, f116, BT, FF, EE, 1);
    gemm_f16<<<gE, blk, 0, stream>>>(f116, w216 + (size_t)l * FF * EE, b2 + l * EE,
                                     x, x, nullptr, BT, EE, FF, 0);
  }

  layernorm_kernel<<<BT / 8, blk, 0, stream>>>(x, lnf_g, lnf_b, h16);

  float* logits = (float*)d_out;
  gemm_f16<<<gH, blk, 0, stream>>>(h16, wh16, bh, nullptr, logits, nullptr, BT, VV, EE, 0);

  loss_rows_kernel<<<BT / 8, blk, 0, stream>>>(logits, targets, nll);
  loss_reduce_kernel<<<1, blk, 0, stream>>>(nll, logits + (size_t)BT * VV);
}
